// StrassenAttention_63093069578813
// MI455X (gfx1250) — matrix operand order repaired and checked by reading, not compiled, not run on hardware
//
#include <hip/hip_runtime.h>

typedef _Float16 v8h  __attribute__((ext_vector_type(8)));
typedef _Float16 v16h __attribute__((ext_vector_type(16)));
typedef float    v8f  __attribute__((ext_vector_type(8)));

#define NTOK   384
#define HEADS  8
#define DHEAD  32
#define CDIM   256
#define SCALER 0.17677669529663687f  /* 1/sqrt(32) */

union FragH { v16h v; v8h h[2]; };

// A-operand fragment (16x32 f16, M x K): lane = M row (lr), halves:
//   lanes 0-15 : K = {k0..k0+7, k0+16..k0+23}
//   lanes 16-31: K = {k0+8..k0+15, k0+24..k0+31}
__device__ __forceinline__ v16h load_a_frag(const _Float16* __restrict__ row, int k0, int hi) {
    FragH f;
    const _Float16* p = row + k0 + (hi ? 8 : 0);
    f.h[0] = *(const v8h*)(p);
    f.h[1] = *(const v8h*)(p + 16);
    return f.v;
}

// B-operand fragment (32x16 f16, K x N), loaded from B^T stored row-major (N x K):
//   lane = N column (lr); lanes 0-15 hold K = {k0..k0+7, k0+16..k0+23}, lanes 16-31 K = {k0+8..k0+15, k0+24..k0+31}
__device__ __forceinline__ v16h load_b_frag(const _Float16* __restrict__ row, int k0, int hi) {
    FragH f;
    const _Float16* p = row + k0 + (hi ? 8 : 0);
    f.h[0] = *(const v8h*)(p);
    f.h[1] = *(const v8h*)(p + 16);
    return f.v;
}

__device__ __forceinline__ v8f wmma_f16(v16h a, v16h b, v8f c) {
    return __builtin_amdgcn_wmma_f32_16x16x32_f16(false, a, false, b, (short)0, c, false, false);
}

// ---------------------------------------------------------------- f32 -> f16
__global__ __launch_bounds__(256) void k_cvt(const float* __restrict__ src,
                                             _Float16* __restrict__ dst, int n) {
    int i = blockIdx.x * 256 + threadIdx.x;
    if (i < n) dst[i] = (_Float16)src[i];
}

// ------------------------------------------------- projection: wx = hid @ W^T
// grid (80, 24): o-tile x n-tile, one wave each. K = 256 -> 8 WMMAs.
// Scatters into per-head a/b/c [t][h][n][d] f16, vjT [h][d][n] f16, vkT [h][d][n] f32.
__global__ __launch_bounds__(32) void k_proj(const _Float16* __restrict__ hidH,
                                             const _Float16* __restrict__ WH,
                                             _Float16* __restrict__ abc,
                                             _Float16* __restrict__ vjT,
                                             float* __restrict__ vkT) {
    const int ot = blockIdx.x, nt = blockIdx.y;
    const int lane = threadIdx.x, hi = lane >> 4, lr = lane & 15;
    const _Float16* arow = hidH + (nt * 16 + lr) * CDIM;
    const _Float16* brow = WH   + (ot * 16 + lr) * CDIM;
    v8f acc = {};
#pragma unroll
    for (int kk = 0; kk < CDIM; kk += 32)
        acc = wmma_f16(load_a_frag(arow, kk, hi), load_b_frag(brow, kk, hi), acc);

    const int o  = ot * 16 + lr;          // output feature for this lane's column
    const int t  = o >> 8;                // which of the 5 projections
    const int hh = (o >> 5) & 7;
    const int dd = o & 31;
    const int n0 = nt * 16 + hi * 8;
#pragma unroll
    for (int r = 0; r < 8; ++r) {
        const float v = acc[r];
        const int n = n0 + r;
        if (t < 3)        abc[((t * HEADS + hh) * NTOK + n) * DHEAD + dd] = (_Float16)v;
        else if (t == 3)  vjT[(hh * DHEAD + dd) * NTOK + n] = (_Float16)v;
        else              vkT[(hh * DHEAD + dd) * NTOK + n] = v;
    }
}

// -------------------------------------- scores: X = b c^T, Yt = a c^T, Z = a b^T
// grid (24, 24, 8). K = D = 32 -> single WMMA per tile per score.
__global__ __launch_bounds__(32) void k_scores(const _Float16* __restrict__ abc,
                                               float* __restrict__ Sx,
                                               float* __restrict__ Syt,
                                               float* __restrict__ Sz) {
    const int t2 = blockIdx.x, t1 = blockIdx.y, h = blockIdx.z;
    const int lane = threadIdx.x, hi = lane >> 4, lr = lane & 15;
    const _Float16* A = abc + (0 * HEADS + h) * NTOK * DHEAD;
    const _Float16* B = abc + (1 * HEADS + h) * NTOK * DHEAD;
    const _Float16* C = abc + (2 * HEADS + h) * NTOK * DHEAD;
    v16h a1 = load_a_frag(A + (t1 * 16 + lr) * DHEAD, 0, hi);
    v16h b1 = load_a_frag(B + (t1 * 16 + lr) * DHEAD, 0, hi);
    v16h b2 = load_b_frag(B + (t2 * 16 + lr) * DHEAD, 0, hi);
    v16h c2 = load_b_frag(C + (t2 * 16 + lr) * DHEAD, 0, hi);
    v8f x = {}, y = {}, z = {};
    x = wmma_f16(b1, c2, x);   // X [j=t1, k=t2]
    y = wmma_f16(a1, c2, y);   // Yt[i=t1, k=t2]
    z = wmma_f16(a1, b2, z);   // Z [i=t1, j=t2]
    const int row0 = t1 * 16 + hi * 8;
    const int col  = t2 * 16 + lr;
    float* px = Sx  + h * NTOK * NTOK;
    float* py = Syt + h * NTOK * NTOK;
    float* pz = Sz  + h * NTOK * NTOK;
#pragma unroll
    for (int r = 0; r < 8; ++r) {
        const int idx = (row0 + r) * NTOK + col;
        px[idx] = x[r] * SCALER;
        py[idx] = y[r] * SCALER;
        pz[idx] = z[r] * SCALER;
    }
}

// ------------------------------------------------------------- row max (384)
__global__ __launch_bounds__(128) void k_rowmax(const float* __restrict__ S,
                                                float* __restrict__ rm) {
    const int row = blockIdx.x;
    const float* p = S + row * NTOK;
    float m = -3.4e38f;
    for (int j = threadIdx.x; j < NTOK; j += 128) m = fmaxf(m, p[j]);
    __shared__ float red[128];
    red[threadIdx.x] = m;
    __syncthreads();
    for (int s = 64; s > 0; s >>= 1) {
        if (threadIdx.x < s) red[threadIdx.x] = fmaxf(red[threadIdx.x], red[threadIdx.x + s]);
        __syncthreads();
    }
    if (threadIdx.x == 0) rm[row] = red[0];
}

// ------------------------------------------ per-head global max of X row-maxes
__global__ __launch_bounds__(128) void k_headmax(const float* __restrict__ rm,
                                                 float* __restrict__ hm) {
    const int h = blockIdx.x;
    const float* p = rm + h * NTOK;
    float m = -3.4e38f;
    for (int j = threadIdx.x; j < NTOK; j += 128) m = fmaxf(m, p[j]);
    __shared__ float red[128];
    red[threadIdx.x] = m;
    __syncthreads();
    for (int s = 64; s > 0; s >>= 1) {
        if (threadIdx.x < s) red[threadIdx.x] = fmaxf(red[threadIdx.x], red[threadIdx.x + s]);
        __syncthreads();
    }
    if (threadIdx.x == 0) hm[h] = red[0];
}

// ------------------- exp(X - headmax), stored TRANSPOSED f16: Ext[h][k][j]
__global__ __launch_bounds__(256) void k_exp_xt(const float* __restrict__ Sx,
                                                const float* __restrict__ hm,
                                                _Float16* __restrict__ Ext) {
    int idx = blockIdx.x * 256 + threadIdx.x;
    if (idx >= HEADS * NTOK * NTOK) return;
    const int h = idx / (NTOK * NTOK);
    const int rem = idx - h * NTOK * NTOK;
    const int k = rem / NTOK, j = rem - k * NTOK;
    Ext[idx] = (_Float16)__expf(Sx[h * NTOK * NTOK + j * NTOK + k] - hm[h]);
}

// ---------------------------- exp(S - rowmax), f32 out (Yt) / f16 out (Z)
__global__ __launch_bounds__(256) void k_exp_row_f32(const float* __restrict__ S,
                                                     const float* __restrict__ rm,
                                                     float* __restrict__ E) {
    int idx = blockIdx.x * 256 + threadIdx.x;
    if (idx >= HEADS * NTOK * NTOK) return;
    E[idx] = __expf(S[idx] - rm[idx / NTOK]);
}
__global__ __launch_bounds__(256) void k_exp_row_f16(const float* __restrict__ S,
                                                     const float* __restrict__ rm,
                                                     _Float16* __restrict__ E) {
    int idx = blockIdx.x * 256 + threadIdx.x;
    if (idx >= HEADS * NTOK * NTOK) return;
    E[idx] = (_Float16)__expf(S[idx] - rm[idx / NTOK]);
}

// ------------------------------------------------------- the cubic contraction
// grid (24, 33, 8) = (i-tile, d [32=Den slice], head), one wave32 per block.
// For d<32: Num[h,i,d] = sum_k Yt[i,k]*vk[k,d]*(Z @ (X .* vj[:,d]))[i,k]
// For d=32: Den[h,i]   = sum_k Yt[i,k]*(Z @ X)[i,k]
__global__ __launch_bounds__(32) void k_big(const _Float16* __restrict__ Ez,
                                            const _Float16* __restrict__ Ext,
                                            const float* __restrict__ Eyt,
                                            const _Float16* __restrict__ vjT,
                                            const float* __restrict__ vkT,
                                            float* __restrict__ Num,
                                            float* __restrict__ Den) {
    const int it = blockIdx.x, d = blockIdx.y, h = blockIdx.z;
    const int lane = threadIdx.x, hi = lane >> 4, lr = lane & 15;
    const int i0 = it * 16;
    const _Float16* Zrow = Ez + h * NTOK * NTOK + (i0 + lr) * NTOK;  // A rows
    const _Float16* Xb   = Ext + h * NTOK * NTOK;                    // B^T rows (k-major)
    const float*    Yb   = Eyt + h * NTOK * NTOK;
    const bool useV = (d < DHEAD);
    const _Float16* vjrow = vjT + (h * DHEAD + (useV ? d : 0)) * NTOK;
    const float*    vkrow = vkT + (h * DHEAD + (useV ? d : 0)) * NTOK;

    float part[8];
#pragma unroll
    for (int r = 0; r < 8; ++r) part[r] = 0.f;

    for (int ktc = 0; ktc < 24; ktc += 4) {   // 4 independent k-tile accumulators
        v8f acc[4] = {};
        for (int jt = 0; jt < 12; ++jt) {
            const int j0 = jt * 32;
            v16h af = load_a_frag(Zrow, j0, hi);
            v16h vf = {};
            if (useV) vf = load_b_frag(vjrow, j0, hi);
#pragma unroll
            for (int c = 0; c < 4; ++c) {
                const int krow = (ktc + c) * 16 + lr;
                v16h bf = load_b_frag(Xb + krow * NTOK, j0, hi);
                if (useV) bf = bf * vf;       // v_pk_mul_f16 row-scale by vj[:,d]
                acc[c] = wmma_f16(af, bf, acc[c]);
            }
        }
        // fused epilogue: weight by Yt[i,k] * vk[k,d], partial-sum over k
#pragma unroll
        for (int c = 0; c < 4; ++c) {
            const int kg = (ktc + c) * 16 + lr;
            const float wv = useV ? vkrow[kg] : 1.0f;
#pragma unroll
            for (int r = 0; r < 8; ++r) {
                const int i = i0 + hi * 8 + r;
                part[r] += acc[c][r] * Yb[i * NTOK + kg] * wv;
            }
        }
    }

    // cross-lane reduction over the 16 k'-stripes
    __shared__ float red[32][8];
#pragma unroll
    for (int r = 0; r < 8; ++r) red[lane][r] = part[r];
    __syncthreads();
    if (lane < 16) {
        const int r = lane & 7;
        const int base = (lane < 8) ? 0 : 16;   // m<8 from lanes 0-15, m>=8 from 16-31
        float s = 0.f;
#pragma unroll
        for (int l = 0; l < 16; ++l) s += red[base + l][r];
        const int i = i0 + lane;
        if (useV) Num[(h * NTOK + i) * DHEAD + d] = s;
        else      Den[h * NTOK + i] = s;
    }
}

// --------------------------------------------- out[n, h*32+d] = Num / (Den+eps)
__global__ __launch_bounds__(256) void k_final(const float* __restrict__ Num,
                                               const float* __restrict__ Den,
                                               float* __restrict__ out) {
    int idx = blockIdx.x * 256 + threadIdx.x;
    if (idx >= NTOK * CDIM) return;
    const int n = idx >> 8, o = idx & 255;
    const int h = o >> 5, d = o & 31;
    out[idx] = Num[(h * NTOK + n) * DHEAD + d] / (Den[h * NTOK + n] + 1e-9f);
}

extern "C" void kernel_launch(void* const* d_in, const int* in_sizes, int n_in,
                              void* d_out, int out_size, void* d_ws, size_t ws_size,
                              hipStream_t stream) {
    (void)in_sizes; (void)n_in; (void)out_size; (void)ws_size;
    const float* hid_f32 = (const float*)d_in[0];   // [384,256]
    const float* W_f32   = (const float*)d_in[1];   // [1280,256]
    float* out = (float*)d_out;

    char* ws = (char*)d_ws;
    size_t off = 0;
    auto alloc = [&](size_t bytes) -> void* {
        void* p = ws + off;
        off += (bytes + 255) & ~(size_t)255;
        return p;
    };
    const size_t NN = (size_t)NTOK * NTOK;          // 147456
    _Float16* hidH = (_Float16*)alloc((size_t)NTOK * CDIM * 2);
    _Float16* WH   = (_Float16*)alloc((size_t)5 * CDIM * CDIM * 2);
    _Float16* abc  = (_Float16*)alloc((size_t)3 * HEADS * NTOK * DHEAD * 2);
    _Float16* vjT  = (_Float16*)alloc((size_t)HEADS * DHEAD * NTOK * 2);
    float*    vkT  = (float*)   alloc((size_t)HEADS * DHEAD * NTOK * 4);
    float*    Sx   = (float*)   alloc(HEADS * NN * 4);
    float*    Syt  = (float*)   alloc(HEADS * NN * 4);
    float*    Sz   = (float*)   alloc(HEADS * NN * 4);
    float*    rmx  = (float*)   alloc((size_t)HEADS * NTOK * 4);
    float*    rmy  = (float*)   alloc((size_t)HEADS * NTOK * 4);
    float*    rmz  = (float*)   alloc((size_t)HEADS * NTOK * 4);
    float*    xmax = (float*)   alloc(HEADS * 4);
    _Float16* Ext  = (_Float16*)alloc(HEADS * NN * 2);
    float*    Eyt  = (float*)   alloc(HEADS * NN * 4);
    _Float16* Ez   = (_Float16*)alloc(HEADS * NN * 2);
    float*    Num  = (float*)   alloc((size_t)HEADS * NTOK * DHEAD * 4);
    float*    Den  = (float*)   alloc((size_t)HEADS * NTOK * 4);

    const int nHid = NTOK * CDIM;        // 98304
    const int nW   = 5 * CDIM * CDIM;    // 327680
    const int nScr = HEADS * (int)NN;    // 1179648

    k_cvt<<<(nHid + 255) / 256, 256, 0, stream>>>(hid_f32, hidH, nHid);
    k_cvt<<<(nW + 255) / 256, 256, 0, stream>>>(W_f32, WH, nW);

    k_proj<<<dim3(80, 24), 32, 0, stream>>>(hidH, WH, abc, vjT, vkT);

    k_scores<<<dim3(24, 24, 8), 32, 0, stream>>>(abc, Sx, Syt, Sz);

    k_rowmax<<<HEADS * NTOK, 128, 0, stream>>>(Sx, rmx);
    k_rowmax<<<HEADS * NTOK, 128, 0, stream>>>(Syt, rmy);
    k_rowmax<<<HEADS * NTOK, 128, 0, stream>>>(Sz, rmz);
    k_headmax<<<HEADS, 128, 0, stream>>>(rmx, xmax);

    k_exp_xt<<<(nScr + 255) / 256, 256, 0, stream>>>(Sx, xmax, Ext);
    k_exp_row_f32<<<(nScr + 255) / 256, 256, 0, stream>>>(Syt, rmy, Eyt);
    k_exp_row_f16<<<(nScr + 255) / 256, 256, 0, stream>>>(Sz, rmz, Ez);

    k_big<<<dim3(24, DHEAD + 1, HEADS), 32, 0, stream>>>(Ez, Ext, Eyt, vjT, vkT, Num, Den);

    k_final<<<(NTOK * CDIM + 255) / 256, 256, 0, stream>>>(Num, Den, out);
}
